// SelfAttentionModule_65481071405622
// MI455X (gfx1250) — matrix operand order repaired and checked by reading, not compiled, not run on hardware
//
#include <hip/hip_runtime.h>

// Problem constants (match reference)
#define B_SZ    4
#define NSEQ    2048
#define DMODEL  1024
#define NHEADS  16
#define HDIM    64
#define MROWS   (B_SZ * NSEQ)   // 8192

typedef _Float16 h16 __attribute__((ext_vector_type(16)));
typedef float    f8  __attribute__((ext_vector_type(8)));

union FragU { uint4 q[2]; h16 h; };

// ---------------------------------------------------------------------------
// WMMA fragment helpers (v_wmma_f32_16x16x32_f16, wave32)
//
// A (16x32 f16): lane = row (l&15); halves 0..7  = K 8*hi .. 8*hi+7
//                                  halves 8..15 = K 16+8*hi .. 23+8*hi
// B (32x16 f16): lane = col (l&15); same K mapping as A
// C (16x16 f32): VGPR r, lane l -> (M = r + 8*(l>>4), N = l&15)
// ---------------------------------------------------------------------------
__device__ __forceinline__ h16 load_a_frag(const _Float16* base, int ld, int lane) {
    int row = lane & 15, hi = lane >> 4;
    const _Float16* p = base + (size_t)row * ld;
    FragU u;
    u.q[0] = *(const uint4*)(p + 8 * hi);
    u.q[1] = *(const uint4*)(p + 16 + 8 * hi);
    return u.h;
}

__device__ __forceinline__ h16 load_b_frag(const _Float16* base, int ld, int lane) {
    int n = lane & 15, hi = lane >> 4;
    const _Float16* p = base + (size_t)n * ld;
    FragU u;
    u.q[0] = *(const uint4*)(p + 8 * hi);
    u.q[1] = *(const uint4*)(p + 16 + 8 * hi);
    return u.h;
}

__device__ __forceinline__ f8 wmma_16x16x32(h16 a, h16 b, f8 c) {
    return __builtin_amdgcn_wmma_f32_16x16x32_f16(false, a, false, b, (short)0, c,
                                                  false, false);
}

// ---------------------------------------------------------------------------
// fp32 -> f16 conversion
// ---------------------------------------------------------------------------
__global__ void cvt_f32_f16_kernel(const float* __restrict__ in,
                                   _Float16* __restrict__ out, int n) {
    int i = blockIdx.x * blockDim.x + threadIdx.x;
    if (i < n) out[i] = (_Float16)in[i];
}

// ---------------------------------------------------------------------------
// Generic WMMA GEMM: C[M x Ncols] = A[M x Kdim] (row major f16) *
//                    W[Ncols x Kdim] (row major f16, i.e. nn.Linear weight)
// 256 threads = 8 waves; wave computes a 32 x 64 C tile:
//   2 A-fragments x 4 B-fragments -> 8 WMMAs per K-step (2x B reuse vs 16x64).
// MODE 0: out16 in [B, H, N, 64]   (Q, K)
// MODE 1: out16 in [B, H, 64, N]   (V transposed)
// MODE 2: out32[m*Ncols+c] = acc + resid  (O projection + residual, fp32)
// ---------------------------------------------------------------------------
template <int MODE>
__global__ void gemm_wmma_kernel(const _Float16* __restrict__ A,
                                 const _Float16* __restrict__ W,
                                 _Float16* __restrict__ out16,
                                 float* __restrict__ out32,
                                 const float* __restrict__ resid,
                                 int Kdim, int Ncols, int Nseq) {
    const int lane = threadIdx.x & 31;
    const int wave = threadIdx.x >> 5;
    const int row0 = (blockIdx.y * 8 + wave) * 32;
    const int col0 = blockIdx.x * 64;

    f8 acc[2][4];
#pragma unroll
    for (int t = 0; t < 2; ++t)
#pragma unroll
        for (int j = 0; j < 4; ++j)
#pragma unroll
            for (int r = 0; r < 8; ++r) acc[t][j][r] = 0.0f;

    for (int k0 = 0; k0 < Kdim; k0 += 32) {
        const _Float16* pA0 = A + (size_t)row0 * Kdim + k0;
        const _Float16* pA1 = A + (size_t)(row0 + 16) * Kdim + k0;
        __builtin_prefetch(pA0 + 32, 0, 1);
        __builtin_prefetch(pA1 + 32, 0, 1);
        h16 a0 = load_a_frag(pA0, Kdim, lane);
        h16 a1 = load_a_frag(pA1, Kdim, lane);
#pragma unroll
        for (int j = 0; j < 4; ++j) {
            h16 b = load_b_frag(W + (size_t)(col0 + 16 * j) * Kdim + k0, Kdim, lane);
            acc[0][j] = wmma_16x16x32(a0, b, acc[0][j]);
            acc[1][j] = wmma_16x16x32(a1, b, acc[1][j]);
        }
    }

    const int colLane = lane & 15, hi = lane >> 4;
#pragma unroll
    for (int t = 0; t < 2; ++t) {
#pragma unroll
        for (int j = 0; j < 4; ++j) {
#pragma unroll
            for (int r = 0; r < 8; ++r) {
                int m = row0 + 16 * t + r + 8 * hi;
                int c = col0 + 16 * j + colLane;
                float v = acc[t][j][r];
                if (MODE == 0) {
                    int b = m / Nseq, n = m % Nseq;
                    int h = c >> 6, d = c & 63;
                    out16[(((size_t)(b * NHEADS + h) * Nseq + n) << 6) + d] =
                        (_Float16)v;
                } else if (MODE == 1) {
                    int b = m / Nseq, n = m % Nseq;
                    int h = c >> 6, d = c & 63;
                    out16[((size_t)(b * NHEADS + h) * HDIM + d) * Nseq + n] =
                        (_Float16)v;
                } else {
                    size_t idx = (size_t)m * Ncols + c;
                    out32[idx] = v + resid[idx];
                }
            }
        }
    }
}

// ---------------------------------------------------------------------------
// Flash attention (online softmax), wave32 WMMA.
// Q, K in [B,H,N,64] f16; Vt in [B,H,64,N] f16; out Ah in [B*N, D] f16.
// 128 threads = 4 waves; wave owns 16 query rows; loop over 32-key blocks.
// ---------------------------------------------------------------------------
__global__ void attn_wmma_kernel(const _Float16* __restrict__ Q,
                                 const _Float16* __restrict__ Kc,
                                 const _Float16* __restrict__ Vt,
                                 _Float16* __restrict__ Ah, int Nseq) {
    const int lane = threadIdx.x & 31;
    const int wave = threadIdx.x >> 5;
    const int colLane = lane & 15, hi = lane >> 4;

    const int qBlocks = Nseq / 64;
    const int bh   = blockIdx.x / qBlocks;
    const int qblk = blockIdx.x % qBlocks;
    const int q0   = qblk * 64 + wave * 16;

    const _Float16* Qbh = Q  + (size_t)bh * Nseq * HDIM;
    const _Float16* Kbh = Kc + (size_t)bh * Nseq * HDIM;
    const _Float16* Vbh = Vt + (size_t)bh * HDIM * Nseq;

    __shared__ __align__(16) _Float16 Pbuf[4][16 * 32];
    _Float16* Pw = &Pbuf[wave][0];

    // Q fragments for the whole Hd=64 (K-dim split into 2x32), kept in regs.
    h16 aq0 = load_a_frag(Qbh + (size_t)q0 * HDIM + 0,  HDIM, lane);
    h16 aq1 = load_a_frag(Qbh + (size_t)q0 * HDIM + 32, HDIM, lane);

    f8 o[4];
    float mrow[8], lrow[8];
#pragma unroll
    for (int j = 0; j < 4; ++j)
#pragma unroll
        for (int r = 0; r < 8; ++r) o[j][r] = 0.0f;
#pragma unroll
    for (int r = 0; r < 8; ++r) { mrow[r] = -1e30f; lrow[r] = 0.0f; }

    const float scale = 0.125f;  // 1/sqrt(64)

    for (int key0 = 0; key0 < Nseq; key0 += 32) {
        // S = scale * Q @ K^T for 32 keys -> two 16x16 C tiles
        f8 S[2];
#pragma unroll
        for (int s = 0; s < 2; ++s) {
            f8 c;
#pragma unroll
            for (int r = 0; r < 8; ++r) c[r] = 0.0f;
            h16 b0 = load_b_frag(Kbh + (size_t)(key0 + 16 * s) * HDIM + 0,  HDIM, lane);
            c = wmma_16x16x32(aq0, b0, c);
            h16 b1 = load_b_frag(Kbh + (size_t)(key0 + 16 * s) * HDIM + 32, HDIM, lane);
            c = wmma_16x16x32(aq1, b1, c);
#pragma unroll
            for (int r = 0; r < 8; ++r) S[s][r] = c[r] * scale;
        }

        // online softmax: row max over 32 keys (16 lanes of our half + 2 tiles)
        float mnew[8], alpha[8];
#pragma unroll
        for (int r = 0; r < 8; ++r) {
            float v = fmaxf(S[0][r], S[1][r]);
#pragma unroll
            for (int off = 1; off < 16; off <<= 1)
                v = fmaxf(v, __shfl_xor(v, off, 32));
            mnew[r]  = fmaxf(mrow[r], v);
            alpha[r] = __expf(mrow[r] - mnew[r]);
            mrow[r]  = mnew[r];
        }
#pragma unroll
        for (int j = 0; j < 4; ++j)
#pragma unroll
            for (int r = 0; r < 8; ++r) o[j][r] *= alpha[r];

        // P = exp(S - m): stage to LDS (C-layout -> A-layout transpose), row sums
#pragma unroll
        for (int r = 0; r < 8; ++r) {
            float p0 = __expf(S[0][r] - mnew[r]);
            float p1 = __expf(S[1][r] - mnew[r]);
            int row = r + 8 * hi;
            Pw[row * 32 + colLane]      = (_Float16)p0;
            Pw[row * 32 + 16 + colLane] = (_Float16)p1;
            float t = p0 + p1;
#pragma unroll
            for (int off = 1; off < 16; off <<= 1)
                t += __shfl_xor(t, off, 32);
            lrow[r] = lrow[r] * alpha[r] + t;
        }
        asm volatile("s_wait_dscnt 0" ::: "memory");

        // A-fragment of P from LDS (ld = 32 halves)
        FragU up;
        up.q[0] = *(const uint4*)(Pw + colLane * 32 + 8 * hi);
        up.q[1] = *(const uint4*)(Pw + colLane * 32 + 16 + 8 * hi);
        h16 pa = up.h;

        // O += P @ V  (Vt row-major [d][n] is the B-layout [N][K])
#pragma unroll
        for (int j = 0; j < 4; ++j) {
            h16 bv = load_b_frag(Vbh + (size_t)(16 * j) * Nseq + key0, Nseq, lane);
            o[j] = wmma_16x16x32(pa, bv, o[j]);
        }
    }

    // normalize and write out as [B*N, D] f16 with col = h*64 + d
    const int b = bh >> 4, h = bh & 15;
    float rl[8];
#pragma unroll
    for (int r = 0; r < 8; ++r) rl[r] = 1.0f / lrow[r];
#pragma unroll
    for (int j = 0; j < 4; ++j)
#pragma unroll
        for (int r = 0; r < 8; ++r) {
            int q = q0 + r + 8 * hi;
            int col = h * HDIM + 16 * j + colLane;
            Ah[(size_t)(b * Nseq + q) * DMODEL + col] = (_Float16)(o[j][r] * rl[r]);
        }
}

// ---------------------------------------------------------------------------
// LayerNorm over D=1024 per row (input already has residual added)
// ---------------------------------------------------------------------------
__global__ void layernorm_kernel(const float* __restrict__ y,
                                 const float* __restrict__ gamma,
                                 const float* __restrict__ beta,
                                 float* __restrict__ out, int D) {
    const int row = blockIdx.x;
    const float* yr = y + (size_t)row * D;
    float s = 0.0f, s2 = 0.0f;
    for (int i = threadIdx.x; i < D; i += blockDim.x) {
        float v = yr[i];
        s += v; s2 += v * v;
    }
    __shared__ float rs[256], rs2[256];
    rs[threadIdx.x] = s; rs2[threadIdx.x] = s2;
    __syncthreads();
    for (int o2 = 128; o2 > 0; o2 >>= 1) {
        if ((int)threadIdx.x < o2) {
            rs[threadIdx.x]  += rs[threadIdx.x + o2];
            rs2[threadIdx.x] += rs2[threadIdx.x + o2];
        }
        __syncthreads();
    }
    float mu   = rs[0] / (float)D;
    float var  = rs2[0] / (float)D - mu * mu;
    float rstd = rsqrtf(var + 1e-5f);
    for (int i = threadIdx.x; i < D; i += blockDim.x) {
        out[(size_t)row * D + i] = (yr[i] - mu) * rstd * gamma[i] + beta[i];
    }
}

// ---------------------------------------------------------------------------
// launch
// ---------------------------------------------------------------------------
extern "C" void kernel_launch(void* const* d_in, const int* in_sizes, int n_in,
                              void* d_out, int out_size, void* d_ws, size_t ws_size,
                              hipStream_t stream) {
    const float* x     = (const float*)d_in[0];
    const float* Wq    = (const float*)d_in[1];
    const float* Wk    = (const float*)d_in[2];
    const float* Wv    = (const float*)d_in[3];
    const float* Wo    = (const float*)d_in[4];
    const float* gamma = (const float*)d_in[5];
    const float* beta  = (const float*)d_in[6];
    float* out = (float*)d_out;

    const size_t MB = 1024ull * 1024ull;
    char* w = (char*)d_ws;
    _Float16* xh  = (_Float16*)(w);             // 16 MB  x in f16
    _Float16* wqh = (_Float16*)(w + 16 * MB);   //  2 MB
    _Float16* wkh = (_Float16*)(w + 18 * MB);   //  2 MB
    _Float16* wvh = (_Float16*)(w + 20 * MB);   //  2 MB
    _Float16* woh = (_Float16*)(w + 22 * MB);   //  2 MB
    _Float16* Qh  = (_Float16*)(w + 24 * MB);   // 16 MB  [B,H,N,64]
    _Float16* Kh  = (_Float16*)(w + 40 * MB);   // 16 MB  [B,H,N,64]
    _Float16* Vth = (_Float16*)(w + 56 * MB);   // 16 MB  [B,H,64,N]
    _Float16* Ah  = (_Float16*)(w + 72 * MB);   // 16 MB  attn out f16 [M,D]
    float*    yv  = (float*)(w + 88 * MB);      // 32 MB  o-proj + residual fp32

    // 1) fp32 -> f16 conversions
    {
        int n = MROWS * DMODEL;
        cvt_f32_f16_kernel<<<(n + 255) / 256, 256, 0, stream>>>(x, xh, n);
        int nw = DMODEL * DMODEL;
        cvt_f32_f16_kernel<<<(nw + 255) / 256, 256, 0, stream>>>(Wq, wqh, nw);
        cvt_f32_f16_kernel<<<(nw + 255) / 256, 256, 0, stream>>>(Wk, wkh, nw);
        cvt_f32_f16_kernel<<<(nw + 255) / 256, 256, 0, stream>>>(Wv, wvh, nw);
        cvt_f32_f16_kernel<<<(nw + 255) / 256, 256, 0, stream>>>(Wo, woh, nw);
    }

    // 2) Q/K/V projections (WMMA GEMM): wave tile 32x64, block covers 256 rows
    dim3 ggrid(DMODEL / 64, MROWS / 256);
    gemm_wmma_kernel<0><<<ggrid, 256, 0, stream>>>(xh, wqh, Qh, nullptr, nullptr,
                                                   DMODEL, DMODEL, NSEQ);
    gemm_wmma_kernel<0><<<ggrid, 256, 0, stream>>>(xh, wkh, Kh, nullptr, nullptr,
                                                   DMODEL, DMODEL, NSEQ);
    gemm_wmma_kernel<1><<<ggrid, 256, 0, stream>>>(xh, wvh, Vth, nullptr, nullptr,
                                                   DMODEL, DMODEL, NSEQ);

    // 3) flash attention
    int ablocks = B_SZ * NHEADS * (NSEQ / 64);
    attn_wmma_kernel<<<ablocks, 128, 0, stream>>>(Qh, Kh, Vth, Ah, NSEQ);

    // 4) output projection + residual (fp32)
    gemm_wmma_kernel<2><<<ggrid, 256, 0, stream>>>(Ah, woh, nullptr, yv, x,
                                                   DMODEL, DMODEL, NSEQ);

    // 5) LayerNorm -> d_out
    layernorm_kernel<<<MROWS, 256, 0, stream>>>(yv, gamma, beta, out, DMODEL);
}
